// CrossModalAttention_11355893530631
// MI455X (gfx1250) — compile-verified, matrix operand order checked by reading, not run on hardware
//
#include <hip/hip_runtime.h>
#include <hip/hip_bf16.h>

typedef unsigned short u16;
typedef unsigned int   u32;
typedef __attribute__((ext_vector_type(4)))  unsigned int u32x4;
typedef __attribute__((ext_vector_type(4)))  int    v4i;
typedef __attribute__((ext_vector_type(8)))  __bf16 v8bf;
typedef __attribute__((ext_vector_type(16))) __bf16 v16bf;
typedef __attribute__((ext_vector_type(8)))  float  v8f;

#define GLOBAL_AS __attribute__((address_space(1)))
#define LDS_AS    __attribute__((address_space(3)))

#define BM 128
#define BN 128
#define BK 64
#define LDSS 72   // padded LDS row stride (ushorts): 144B = 9*16B -> b128-aligned, conflict-free frag reads
#define NTHR 256

// ---- CDNA5 async global->LDS path (guarded; falls back to VGPR-staged copies) ----
#if defined(__has_builtin)
#  if __has_builtin(__builtin_amdgcn_global_load_async_to_lds_b128)
#    define ASYNC_LDS 1
#  endif
#endif
#ifndef ASYNC_LDS
#  define ASYNC_LDS 0
#endif

__device__ __forceinline__ void wait_async_lds(){
#if defined(__has_builtin)
#  if __has_builtin(__builtin_amdgcn_s_wait_asynccnt)
  __builtin_amdgcn_s_wait_asynccnt(0);
  return;
#  endif
#endif
  asm volatile("s_wait_asynccnt 0x0" ::: "memory");
}

__device__ __forceinline__ u16 f2bf(float f){
  union { float f; u32 u; } v; v.f = f;
  u32 u = v.u;
  return (u16)((u + 0x7FFFu + ((u >> 16) & 1u)) >> 16);  // round-to-nearest-even
}
__device__ __forceinline__ float bf2f(u16 h){
  union { u32 u; float f; } v; v.u = ((u32)h) << 16;
  return v.f;
}

__global__ void f32_to_bf16_kernel(const float* __restrict__ in, u16* __restrict__ out, int n){
  int i = blockIdx.x * blockDim.x + threadIdx.x;
  int stride = gridDim.x * blockDim.x;
  for (; i < n; i += stride) out[i] = f2bf(in[i]);
}

// C[M,N] = A[M,K](bf16 row-major, lda) * Bw[N,K](bf16 row-major, ldb)^T + bias[N]
// EPI: 0 = bf16 out (bias); 1 = bf16 out (bias + exact GELU); 2 = f32 out (bias + bf16 residuals R1,R2)
template<int EPI>
__global__ __launch_bounds__(NTHR)
void gemm_bf16_wmma(const u16* __restrict__ A, int lda,
                    const u16* __restrict__ Bw, int ldb,
                    const float* __restrict__ bias,
                    void* __restrict__ Cout, int ldc,
                    const u16* __restrict__ R1, const u16* __restrict__ R2, int ldr,
                    int K)
{
  __shared__ u16 As[2][BM * LDSS];
  __shared__ u16 Bs[2][BN * LDSS];

  const int t      = threadIdx.x;
  const int lane   = t & 31;
  const int laneMN = lane & 15;   // M row (A frag) / N col (B frag) within 16x16
  const int laneHi = lane >> 4;
  const int laneK  = laneHi * 8;  // lanes 0-15: K 0-7 & 16-23 ; lanes 16-31: K 8-15 & 24-31
  const int wave   = t >> 5;
  const int waveM  = (wave >> 2) * 64;
  const int waveN  = (wave & 3) * 32;

  const int blockM = blockIdx.y * BM;
  const int blockN = blockIdx.x * BN;

  // global->LDS fill: each thread covers 32 contiguous bf16 (four 16B chunks) of one tile row
  const int fRow = t >> 1;          // 0..127
  const int fCol = (t & 1) * 32;    // 0 or 32
  const u16* gA = A  + (size_t)(blockM + fRow) * lda + fCol;
  const u16* gB = Bw + (size_t)(blockN + fRow) * ldb + fCol;
  const int fOff = fRow * LDSS + fCol;

  v8f acc[4][2] = {};

#if ASYNC_LDS
  // direct global->LDS async copies (ASYNCcnt-tracked)
  auto fill = [&](int buf, int kt){
    const u16* pa = gA + (size_t)kt * BK;
    const u16* pb = gB + (size_t)kt * BK;
#pragma unroll
    for (int c = 0; c < 4; ++c){
      __builtin_amdgcn_global_load_async_to_lds_b128(
          (GLOBAL_AS v4i*)(pa + c * 8),
          (LDS_AS v4i*)&As[buf][fOff + c * 8], 0, 0);
      __builtin_amdgcn_global_load_async_to_lds_b128(
          (GLOBAL_AS v4i*)(pb + c * 8),
          (LDS_AS v4i*)&Bs[buf][fOff + c * 8], 0, 0);
    }
  };
  fill(0, 0);
  wait_async_lds();
  __syncthreads();
#else
  u32x4 ra[4], rb[4];
  auto loadRegs = [&](int kt){
    const u16* pa = gA + (size_t)kt * BK;
    const u16* pb = gB + (size_t)kt * BK;
#pragma unroll
    for (int c = 0; c < 4; ++c){
      ra[c] = *(const u32x4*)(pa + c * 8);
      rb[c] = *(const u32x4*)(pb + c * 8);
    }
  };
  auto storeLDS = [&](int buf){
#pragma unroll
    for (int c = 0; c < 4; ++c){
      *(u32x4*)&As[buf][fOff + c * 8] = ra[c];
      *(u32x4*)&Bs[buf][fOff + c * 8] = rb[c];
    }
  };
  loadRegs(0);
  storeLDS(0);
  __syncthreads();
#endif

  const int nk = K / BK;
  for (int kt = 0; kt < nk; ++kt){
    const int buf  = kt & 1;
    const bool more = (kt + 1) < nk;

#if ASYNC_LDS
    if (more) fill(buf ^ 1, kt + 1);   // async into the other buffer while we compute
#else
    if (more) loadRegs(kt + 1);
#endif

#pragma unroll
    for (int kk = 0; kk < 2; ++kk){
      v16bf af[4], bfr[2];
#pragma unroll
      for (int i = 0; i < 4; ++i){
        const u16* p = &As[buf][(waveM + i * 16 + laneMN) * LDSS + kk * 32 + laneK];
        v8bf lo = *(const v8bf*)p;
        v8bf hi = *(const v8bf*)(p + 16);
        af[i] = __builtin_shufflevector(lo, hi, 0,1,2,3,4,5,6,7,8,9,10,11,12,13,14,15);
      }
#pragma unroll
      for (int j = 0; j < 2; ++j){
        const u16* p = &Bs[buf][(waveN + j * 16 + laneMN) * LDSS + kk * 32 + laneK];
        v8bf lo = *(const v8bf*)p;
        v8bf hi = *(const v8bf*)(p + 16);
        bfr[j] = __builtin_shufflevector(lo, hi, 0,1,2,3,4,5,6,7,8,9,10,11,12,13,14,15);
      }
#pragma unroll
      for (int i = 0; i < 4; ++i)
#pragma unroll
        for (int j = 0; j < 2; ++j)
          acc[i][j] = __builtin_amdgcn_wmma_f32_16x16x32_bf16(
              false, af[i], false, bfr[j], (short)0, acc[i][j], false, false);
    }

#if ASYNC_LDS
    if (more) wait_async_lds();
#else
    if (more) storeLDS(buf ^ 1);
#endif
    __syncthreads();
  }

  // ---- epilogue: lane holds column N=laneMN; rows M = e + 8*laneHi ----
  float bval[2];
#pragma unroll
  for (int j = 0; j < 2; ++j)
    bval[j] = bias[blockN + waveN + j * 16 + laneMN];

#pragma unroll
  for (int i = 0; i < 4; ++i){
    const int mbase = blockM + waveM + i * 16 + laneHi * 8;
#pragma unroll
    for (int j = 0; j < 2; ++j){
      const int ncol = blockN + waveN + j * 16 + laneMN;
#pragma unroll
      for (int e = 0; e < 8; ++e){
        const int r = mbase + e;
        float v = acc[i][j][e] + bval[j];
        if (EPI == 1){
          v = 0.5f * v * (1.0f + erff(v * 0.70710678118654752f));  // exact GELU
        }
        if (EPI == 2){
          const size_t ro = (size_t)r * ldr + ncol;
          v += bf2f(R1[ro]) + bf2f(R2[ro]);
          ((float*)Cout)[(size_t)r * ldc + ncol] = v;
        } else {
          ((u16*)Cout)[(size_t)r * ldc + ncol] = f2bf(v);
        }
      }
    }
  }
}

extern "C" void kernel_launch(void* const* d_in, const int* in_sizes, int n_in,
                              void* d_out, int out_size, void* d_ws, size_t ws_size,
                              hipStream_t stream)
{
  (void)in_sizes; (void)n_in; (void)out_size;

  const float* Xg = (const float*)d_in[0];
  const float* Xt = (const float*)d_in[1];
  const float* Wg = (const float*)d_in[2];
  const float* bg = (const float*)d_in[3];
  const float* Wt = (const float*)d_in[4];
  const float* bt = (const float*)d_in[5];
  const float* Wv = (const float*)d_in[6];
  const float* bv = (const float*)d_in[7];
  const float* Wo = (const float*)d_in[8];
  const float* bo = (const float*)d_in[9];
  const float* W1 = (const float*)d_in[10];
  const float* b1 = (const float*)d_in[11];
  const float* W2 = (const float*)d_in[12];
  const float* b2 = (const float*)d_in[13];

  const int Bm = 16384, F = 1024, GK = 512, TK = 768;

  char* w = (char*)d_ws;
  size_t used = 0;
  auto carve = [&](size_t elems)->u16*{
    u16* p = (u16*)(w + used);
    used += (elems * sizeof(u16) + 255) & ~(size_t)255;
    return p;
  };
  u16* bXg = carve((size_t)Bm * GK);
  u16* bXt = carve((size_t)Bm * TK);
  u16* bWg = carve((size_t)F * GK);
  u16* bWt = carve((size_t)F * TK);
  u16* bWv = carve((size_t)F * F);
  u16* bWo = carve((size_t)F * F);
  u16* bW1 = carve((size_t)F * 2 * F);
  u16* bW2 = carve((size_t)F * F);
  u16* G   = carve((size_t)Bm * F);
  u16* T   = carve((size_t)Bm * F);
  u16* V1  = carve((size_t)Bm * F);
  u16* V2  = carve((size_t)Bm * F);
  u16* Cat = carve((size_t)Bm * 2 * F);
  u16* H   = carve((size_t)Bm * F);
  if (used > ws_size) return;  // workspace too small: bail, no OOB writes

  auto conv = [&](const float* s, u16* d, size_t n){
    int thr = 256;
    long long blk = ((long long)n + thr - 1) / thr;
    if (blk > 8192) blk = 8192;
    f32_to_bf16_kernel<<<dim3((unsigned)blk), dim3(thr), 0, stream>>>(s, d, (int)n);
  };
  conv(Xg, bXg, (size_t)Bm * GK);
  conv(Xt, bXt, (size_t)Bm * TK);
  conv(Wg, bWg, (size_t)F * GK);
  conv(Wt, bWt, (size_t)F * TK);
  conv(Wv, bWv, (size_t)F * F);
  conv(Wo, bWo, (size_t)F * F);
  conv(W1, bW1, (size_t)F * 2 * F);
  conv(W2, bW2, (size_t)F * F);

  const dim3 grid(F / BN, Bm / BM);   // 8 x 128
  const dim3 blk(NTHR);

  // 1-2: input projections
  gemm_bf16_wmma<0><<<grid, blk, 0, stream>>>(bXg, GK, bWg, GK, bg, G, F, nullptr, nullptr, 0, GK);
  gemm_bf16_wmma<0><<<grid, blk, 0, stream>>>(bXt, TK, bWt, TK, bt, T, F, nullptr, nullptr, 0, TK);
  // 3-4: value projections (degenerate 1-token attention)
  gemm_bf16_wmma<0><<<grid, blk, 0, stream>>>(T, F, bWv, F, bv, V1, F, nullptr, nullptr, 0, F);
  gemm_bf16_wmma<0><<<grid, blk, 0, stream>>>(G, F, bWv, F, bv, V2, F, nullptr, nullptr, 0, F);
  // 5-6: out projections, written into concat buffer (free concat via ldc=2F)
  gemm_bf16_wmma<0><<<grid, blk, 0, stream>>>(V1, F, bWo, F, bo, Cat,     2 * F, nullptr, nullptr, 0, F);
  gemm_bf16_wmma<0><<<grid, blk, 0, stream>>>(V2, F, bWo, F, bo, Cat + F, 2 * F, nullptr, nullptr, 0, F);
  // 7: fusion layer 1 + exact GELU (K = 2048)
  gemm_bf16_wmma<1><<<grid, blk, 0, stream>>>(Cat, 2 * F, bW1, 2 * F, b1, H, F, nullptr, nullptr, 0, 2 * F);
  // 8: fusion layer 2 + residual (gnn_proj + tr_proj), fp32 output
  gemm_bf16_wmma<2><<<grid, blk, 0, stream>>>(H, F, bW2, F, b2, d_out, F, G, T, F, F);
}
